// GATModel_22110491640032
// MI455X (gfx1250) — compile-verified, matrix operand order checked by reading, not run on hardware
//
#include <hip/hip_runtime.h>
#include <hip/hip_bf16.h>

typedef __attribute__((ext_vector_type(16))) _Float16 v16h;
typedef __attribute__((ext_vector_type(8)))  _Float16 v8h;
typedef __attribute__((ext_vector_type(8)))  float    v8f;

// ---------------------------------------------------------------------------
// helpers: order-preserving uint encoding for float atomic max
// ---------------------------------------------------------------------------
__device__ __forceinline__ unsigned f2ord(float f) {
    unsigned b = __float_as_uint(f);
    return (b & 0x80000000u) ? ~b : (b | 0x80000000u);
}
__device__ __forceinline__ float ord2f(unsigned u) {
    return (u & 0x80000000u) ? __uint_as_float(u & 0x7FFFFFFFu)
                             : __uint_as_float(~u);
}
#define ORD_NEG_INF 0x007FFFFFu  // f2ord(-inf)

// ---------------------------------------------------------------------------
// utility kernels
// ---------------------------------------------------------------------------
__global__ void fill_u32_kernel(unsigned* __restrict__ p, unsigned v, size_t n) {
    size_t i = (size_t)blockIdx.x * blockDim.x + threadIdx.x;
    if (i < n) p[i] = v;
}

__global__ void f32_to_f16_kernel(const float* __restrict__ in,
                                  _Float16* __restrict__ out, size_t n) {
    size_t i = (size_t)blockIdx.x * blockDim.x + threadIdx.x;
    if (i < n) out[i] = (_Float16)in[i];
}

// ---------------------------------------------------------------------------
// Pack row-major f32 weights W[K][Nc] into per-lane WMMA B fragments (f16).
// Fragment layout (wave32, 16x16x32): lane holds column nt*16 + (lane&15),
// K-halves selected by lane>>4: K = kt*32 + hl*8 + {0..7, 16..23}.
// Packed index of (kt, nt, lane, j) = (((kt*ntN + nt)*32 + lane)*16 + j).
// One thread per (kt, nt, lane) writes 16 contiguous halves (32 B).
// ---------------------------------------------------------------------------
__global__ void pack_w_kernel(const float* __restrict__ W,
                              _Float16* __restrict__ Bp, int K, int Nc) {
    int ntN = Nc >> 4;
    int nkt = K >> 5;
    int t = blockIdx.x * blockDim.x + threadIdx.x;
    if (t >= nkt * ntN * 32) return;
    int lane = t & 31;
    int rest = t >> 5;
    int nt   = rest % ntN;
    int kt   = rest / ntN;
    int hl = lane >> 4, l15 = lane & 15;
    int col = nt * 16 + l15;
    int kbase = kt * 32 + hl * 8;
    _Float16* out = Bp + (size_t)t * 16;
#pragma unroll
    for (int j = 0; j < 8; ++j) {
        out[j]     = (_Float16)W[(size_t)(kbase + j) * Nc + col];
        out[j + 8] = (_Float16)W[(size_t)(kbase + 16 + j) * Nc + col];
    }
}

// ---------------------------------------------------------------------------
// WMMA GEMM:  C[M x Nc] = A[M x K] * Bp   (A f16 row-major, Bp packed frags)
// One wave32 computes a 16 x 64 output strip: 4 N-tiles, 4 accumulators,
// A fragment reused across 4 WMMAs per K-step.  Requires Nc % 64 == 0,
// K % 32 == 0, M % 16 == 0 (here M = 50000 = 3125*16, K = 128, Nc = 128|64).
// ---------------------------------------------------------------------------
__global__ void gemm_wmma_f16_kernel(const _Float16* __restrict__ A,
                                     const _Float16* __restrict__ Bp,
                                     float* __restrict__ C,
                                     int M, int K, int Nc) {
    const int NTB = 4;                      // N-tiles per wave
    int ntN  = Nc >> 4;
    int ngrp = ntN / NTB;                   // wave-groups per row tile
    int wave = (int)((blockIdx.x * blockDim.x + threadIdx.x) >> 5);
    int lane = threadIdx.x & 31;
    int mt = wave / ngrp;
    int ng = wave - mt * ngrp;
    if (mt * 16 >= M) return;
    int m0  = mt * 16;
    int nt0 = ng * NTB;

    int hl = lane >> 4, l15 = lane & 15;
    v8f acc0 = {}, acc1 = {}, acc2 = {}, acc3 = {};
    int nkt = K >> 5;
    for (int kt = 0; kt < nkt; ++kt) {
        // A fragment: two contiguous 16-B vector loads
        const _Float16* ap = A + (size_t)(m0 + l15) * K + kt * 32 + hl * 8;
        v8h alo = *(const v8h*)ap;
        v8h ahi = *(const v8h*)(ap + 16);
        v16h a;
#pragma unroll
        for (int j = 0; j < 8; ++j) { a[j] = alo[j]; a[j + 8] = ahi[j]; }
        // B fragments: contiguous 32-B vector loads from packed layout;
        // consecutive N-tiles are 32 v16h units apart.
        const v16h* bp = (const v16h*)Bp + ((size_t)kt * ntN + nt0) * 32 + lane;
        v16h b0 = bp[0];
        v16h b1 = bp[32];
        v16h b2 = bp[64];
        v16h b3 = bp[96];
        acc0 = __builtin_amdgcn_wmma_f32_16x16x32_f16(false, a, false, b0, (short)0, acc0, false, false);
        acc1 = __builtin_amdgcn_wmma_f32_16x16x32_f16(false, a, false, b1, (short)0, acc1, false, false);
        acc2 = __builtin_amdgcn_wmma_f32_16x16x32_f16(false, a, false, b2, (short)0, acc2, false, false);
        acc3 = __builtin_amdgcn_wmma_f32_16x16x32_f16(false, a, false, b3, (short)0, acc3, false, false);
    }
    // D layout: VGPR v -> row m0 + v + hl*8, col nt*16 + l15
    float* cbase = C + (size_t)(m0 + hl * 8) * Nc + l15;
#pragma unroll
    for (int v = 0; v < 8; ++v) {
        float* cr = cbase + (size_t)v * Nc;
        cr[(nt0 + 0) * 16] = acc0[v];
        cr[(nt0 + 1) * 16] = acc1[v];
        cr[(nt0 + 2) * 16] = acc2[v];
        cr[(nt0 + 3) * 16] = acc3[v];
    }
}

// ---------------------------------------------------------------------------
// per-(node,head) attention scores: s_src = <xh_row, a_src>, s_dst likewise
// ---------------------------------------------------------------------------
__global__ void node_scores_kernel(const float* __restrict__ xh,
                                   const float* __restrict__ a_src,
                                   const float* __restrict__ a_dst,
                                   float* __restrict__ s_src,
                                   float* __restrict__ s_dst,
                                   int n, int heads, int C) {
    int i = blockIdx.x * blockDim.x + threadIdx.x;
    if (i >= n * heads) return;
    int node = i / heads, h = i - node * heads;
    const float* xr = xh + ((size_t)node * heads + h) * C;
    const float* as = a_src + h * C;
    const float* ad = a_dst + h * C;
    float ss = 0.f, sd = 0.f;
    for (int c = 0; c < C; ++c) { float v = xr[c]; ss += v * as[c]; sd += v * ad[c]; }
    s_src[i] = ss;
    s_dst[i] = sd;
}

// ---------------------------------------------------------------------------
// pass 1: e = leaky_relu(s_src[src]+s_dst[dst]); store; atomic segment-max
// ---------------------------------------------------------------------------
__global__ void edge_logits_max_kernel(const int* __restrict__ ei, int E, int Etot,
                                       const float* __restrict__ s_src,
                                       const float* __restrict__ s_dst,
                                       float* __restrict__ elog,
                                       unsigned* __restrict__ mkey, int heads) {
    int i = blockIdx.x * blockDim.x + threadIdx.x;
    if (i >= Etot) return;
    int s, d;
    if (i < E) { s = ei[i]; d = ei[E + i]; } else { s = d = i - E; }
    for (int h = 0; h < heads; ++h) {
        float e = s_src[s * heads + h] + s_dst[d * heads + h];
        e = (e > 0.f) ? e : 0.2f * e;
        elog[(size_t)i * heads + h] = e;
        atomicMax(&mkey[d * heads + h], f2ord(e));
    }
}

// ---------------------------------------------------------------------------
// pass 2: p = exp(e - max[dst]); store p over e; atomic segment-sum denom
// ---------------------------------------------------------------------------
__global__ void edge_expsum_kernel(const int* __restrict__ ei, int E, int Etot,
                                   float* __restrict__ elog,
                                   const unsigned* __restrict__ mkey,
                                   float* __restrict__ denom, int heads) {
    int i = blockIdx.x * blockDim.x + threadIdx.x;
    if (i >= Etot) return;
    int d = (i < E) ? ei[E + i] : (i - E);
    for (int h = 0; h < heads; ++h) {
        float m = ord2f(mkey[d * heads + h]);
        float p = __expf(elog[(size_t)i * heads + h] - m);
        elog[(size_t)i * heads + h] = p;
        atomicAdd(&denom[d * heads + h], p);
    }
}

// ---------------------------------------------------------------------------
// pass 3: out[dst] += alpha * xh[src].  One wave32 per (edge,head);
// 64 columns -> lanes cover c = lane and lane+32 (coalesced atomic adds,
// working set L2-resident on MI455X's 192 MB L2).
// ---------------------------------------------------------------------------
__global__ void edge_aggregate_kernel(const int* __restrict__ ei, int E, int Etot,
                                      const float* __restrict__ elog,
                                      const float* __restrict__ denom,
                                      const float* __restrict__ xh,
                                      float* __restrict__ out, int heads) {
    int w    = (int)((blockIdx.x * blockDim.x + threadIdx.x) >> 5);
    int lane = threadIdx.x & 31;
    int total = Etot * heads;
    if (w >= total) return;
    int i = w / heads, h = w - i * heads;
    int s, d;
    if (i < E) { s = ei[i]; d = ei[E + i]; } else { s = d = i - E; }
    float p = elog[(size_t)i * heads + h];
    float a = p / (denom[d * heads + h] + 1e-16f);
    const float* xs = xh + ((size_t)s * heads + h) * 64;
    float*       od = out + ((size_t)d * heads + h) * 64;
    atomicAdd(&od[lane],      a * xs[lane]);
    atomicAdd(&od[lane + 32], a * xs[lane + 32]);
}

// ---------------------------------------------------------------------------
// h = relu(acc + bias); also emit f16 copy for next WMMA GEMM
// ---------------------------------------------------------------------------
__global__ void relu_bias_f16_kernel(float* __restrict__ acc,
                                     const float* __restrict__ bias,
                                     _Float16* __restrict__ out16,
                                     int n, int F) {
    size_t i = (size_t)blockIdx.x * blockDim.x + threadIdx.x;
    if (i >= (size_t)n * F) return;
    float v = acc[i] + bias[i % F];
    v = (v > 0.f) ? v : 0.f;
    acc[i] = v;
    out16[i] = (_Float16)v;
}

// ---------------------------------------------------------------------------
// final: h2 = relu(acc2 + b2); mortality = h2@Wm+bm; los = h2@Wl+bl
// ---------------------------------------------------------------------------
__global__ void final_heads_kernel(const float* __restrict__ acc2,
                                   const float* __restrict__ b2,
                                   const float* __restrict__ Wm,
                                   const float* __restrict__ bm,
                                   const float* __restrict__ Wl,
                                   const float* __restrict__ bl,
                                   float* __restrict__ outMort,
                                   float* __restrict__ outLos, int n) {
    int i = blockIdx.x * blockDim.x + threadIdx.x;
    if (i >= n) return;
    const float* r = acc2 + (size_t)i * 64;
    float m = 0.f, l0 = 0.f, l1 = 0.f, l2 = 0.f, l3 = 0.f;
    for (int c = 0; c < 64; ++c) {
        float v = r[c] + b2[c];
        v = (v > 0.f) ? v : 0.f;
        m  += v * Wm[c];
        l0 += v * Wl[c * 4 + 0];
        l1 += v * Wl[c * 4 + 1];
        l2 += v * Wl[c * 4 + 2];
        l3 += v * Wl[c * 4 + 3];
    }
    outMort[i] = m + bm[0];
    float* lp = outLos + (size_t)i * 4;
    lp[0] = l0 + bl[0]; lp[1] = l1 + bl[1];
    lp[2] = l2 + bl[2]; lp[3] = l3 + bl[3];
}

// ---------------------------------------------------------------------------
// launcher
// ---------------------------------------------------------------------------
extern "C" void kernel_launch(void* const* d_in, const int* in_sizes, int n_in,
                              void* d_out, int out_size, void* d_ws, size_t ws_size,
                              hipStream_t stream) {
    const float* x      = (const float*)d_in[0];
    const int*   ei     = (const int*)  d_in[1];
    const float* W1     = (const float*)d_in[2];
    const float* a_src1 = (const float*)d_in[3];
    const float* a_dst1 = (const float*)d_in[4];
    const float* b1     = (const float*)d_in[5];
    const float* W2     = (const float*)d_in[6];
    const float* a_src2 = (const float*)d_in[7];
    const float* a_dst2 = (const float*)d_in[8];
    const float* b2     = (const float*)d_in[9];
    const float* Wm     = (const float*)d_in[10];
    const float* bm     = (const float*)d_in[11];
    const float* Wl     = (const float*)d_in[12];
    const float* bl     = (const float*)d_in[13];

    const int F_IN = 128, HID = 64, H1 = 2;
    const int F1   = H1 * HID;                 // 128
    const int N    = in_sizes[0] / F_IN;       // 50000
    const int E    = in_sizes[1] / 2;          // 1600000
    const int Etot = E + N;

    // ---- workspace carve-out (explicit lifetimes, regions reused) ----
    size_t off = 0;
    char* ws = (char*)d_ws;
    auto take = [&](size_t bytes) -> char* {
        char* p = ws + off;
        off += (bytes + 255) & ~(size_t)255;
        return p;
    };
    _Float16* W1p   = (_Float16*)take((size_t)F_IN * F1 * 2);  // packed frags
    _Float16* W2p   = (_Float16*)take((size_t)F1 * HID * 2);   // packed frags
    _Float16* bufA  = (_Float16*)take((size_t)N * 128 * 2);    // x16, then h16
    float*    bufB  = (float*)   take((size_t)N * 128 * 4);    // xh1, then xh2
    float*    bufC  = (float*)   take((size_t)N * 128 * 4);    // out1/h, then out2
    float*    elog  = (float*)   take((size_t)Etot * H1 * 4);
    float*    sSrc1 = (float*)   take((size_t)N * H1 * 4);
    float*    sDst1 = (float*)   take((size_t)N * H1 * 4);
    unsigned* m1    = (unsigned*)take((size_t)N * H1 * 4);
    float*    dn1   = (float*)   take((size_t)N * H1 * 4);
    float*    sSrc2 = (float*)   take((size_t)N * 4);
    float*    sDst2 = (float*)   take((size_t)N * 4);
    unsigned* m2    = (unsigned*)take((size_t)N * 4);
    float*    dn2   = (float*)   take((size_t)N * 4);
    (void)ws_size; (void)n_in; (void)out_size;

    const int BT = 256;
    auto blocks1d = [](size_t n, int bt) { return (unsigned)((n + bt - 1) / bt); };

    // ---- weight packing + input conversion ----
    pack_w_kernel<<<blocks1d((size_t)(F_IN / 32) * (F1 / 16) * 32, BT), BT, 0, stream>>>(W1, W1p, F_IN, F1);
    pack_w_kernel<<<blocks1d((size_t)(F1 / 32) * (HID / 16) * 32, BT), BT, 0, stream>>>(W2, W2p, F1, HID);
    f32_to_f16_kernel<<<blocks1d((size_t)N * F_IN, BT), BT, 0, stream>>>(x, bufA, (size_t)N * F_IN);

    // ---- layer 1 accumulator init ----
    fill_u32_kernel<<<blocks1d((size_t)N * H1, BT), BT, 0, stream>>>(m1, ORD_NEG_INF, (size_t)N * H1);
    fill_u32_kernel<<<blocks1d((size_t)N * H1, BT), BT, 0, stream>>>((unsigned*)dn1, 0u, (size_t)N * H1);
    fill_u32_kernel<<<blocks1d((size_t)N * F1, BT), BT, 0, stream>>>((unsigned*)bufC, 0u, (size_t)N * F1);

    // ---- GEMM1: xh1 = x @ W1  (WMMA, 16x64 strip per wave) ----
    {
        size_t waves = (size_t)(N / 16) * (F1 / 64);
        gemm_wmma_f16_kernel<<<blocks1d(waves * 32, BT), BT, 0, stream>>>(bufA, W1p, bufB, N, F_IN, F1);
    }
    node_scores_kernel<<<blocks1d((size_t)N * H1, BT), BT, 0, stream>>>(bufB, a_src1, a_dst1, sSrc1, sDst1, N, H1, HID);

    // ---- layer 1 edge softmax + aggregation ----
    edge_logits_max_kernel<<<blocks1d(Etot, BT), BT, 0, stream>>>(ei, E, Etot, sSrc1, sDst1, elog, m1, H1);
    edge_expsum_kernel<<<blocks1d(Etot, BT), BT, 0, stream>>>(ei, E, Etot, elog, m1, dn1, H1);
    edge_aggregate_kernel<<<blocks1d((size_t)Etot * H1 * 32, BT), BT, 0, stream>>>(ei, E, Etot, elog, dn1, bufB, bufC, H1);

    // ---- h = relu(out1 + b1), pack f16 for GEMM2 ----
    relu_bias_f16_kernel<<<blocks1d((size_t)N * F1, BT), BT, 0, stream>>>(bufC, b1, bufA, N, F1);

    // ---- GEMM2: xh2 = h @ W2 (WMMA), overwrites xh1 (dead) ----
    {
        size_t waves = (size_t)(N / 16) * (HID / 64);
        gemm_wmma_f16_kernel<<<blocks1d(waves * 32, BT), BT, 0, stream>>>(bufA, W2p, bufB, N, F1, HID);
    }
    node_scores_kernel<<<blocks1d((size_t)N, BT), BT, 0, stream>>>(bufB, a_src2, a_dst2, sSrc2, sDst2, N, 1, HID);

    // ---- layer 2 accumulator init (out2 reuses bufC; h is dead now) ----
    fill_u32_kernel<<<blocks1d((size_t)N, BT), BT, 0, stream>>>(m2, ORD_NEG_INF, (size_t)N);
    fill_u32_kernel<<<blocks1d((size_t)N, BT), BT, 0, stream>>>((unsigned*)dn2, 0u, (size_t)N);
    fill_u32_kernel<<<blocks1d((size_t)N * HID, BT), BT, 0, stream>>>((unsigned*)bufC, 0u, (size_t)N * HID);

    // ---- layer 2 edge softmax + aggregation ----
    edge_logits_max_kernel<<<blocks1d(Etot, BT), BT, 0, stream>>>(ei, E, Etot, sSrc2, sDst2, elog, m2, 1);
    edge_expsum_kernel<<<blocks1d(Etot, BT), BT, 0, stream>>>(ei, E, Etot, elog, m2, dn2, 1);
    edge_aggregate_kernel<<<blocks1d((size_t)Etot * 32, BT), BT, 0, stream>>>(ei, E, Etot, elog, dn2, bufB, bufC, 1);

    // ---- final heads: relu(out2+b2) -> mortality [N], los [N,4] ----
    float* outMort = (float*)d_out;
    float* outLos  = (float*)d_out + N;
    final_heads_kernel<<<blocks1d((size_t)N, BT), BT, 0, stream>>>(bufC, b2, Wm, bm, Wl, bl, outMort, outLos, N);
}
